// AHardPair_20572893348487
// MI455X (gfx1250) — matrix operand order repaired and checked by reading, not compiled, not run on hardware
//
#include <hip/hip_runtime.h>
#include <hip/hip_bf16.h>
#include <math.h>

typedef __attribute__((ext_vector_type(16))) _Float16 v16h;
typedef __attribute__((ext_vector_type(8)))  float    v8f;

#define NPTS 8192
#define DIM  128
#define ALPHA 40.0f
#define BETA  20.0f
#define L2E   1.4426950408889634f

// Packed B-layout index: for column `col`, K index `k`, the element lives at
//   tile ct = col/16, chunk c = k/32, lane = (k%32 / 16)*16 + col%16, slot = k%16
// Linear: (((ct*4 + c)*32 + lane)*16 + slot)
__device__ __forceinline__ size_t pack_idx(int col, int k) {
    size_t ct = (size_t)(col >> 4);
    size_t c  = (size_t)(k >> 5);
    size_t ln = (size_t)(((k >> 4) & 1) * 16 + (col & 15));
    return ((ct * 4 + c) * 32 + ln) * 16 + (size_t)(k & 15);
}

// ---------------------------------------------------------------------------
// Kernel 0a: per-row squared norms (matches reference's sq = sum(x*x))
// ---------------------------------------------------------------------------
__global__ __launch_bounds__(256) void pairloss_norms(const float* __restrict__ x,
                                                      float* __restrict__ norms) {
    int i = blockIdx.x * blockDim.x + threadIdx.x;
    if (i >= NPTS) return;
    const float4* p = (const float4*)(x + (size_t)i * DIM);
    float s = 0.0f;
#pragma unroll 8
    for (int k = 0; k < DIM / 4; ++k) {
        float4 v = p[k];
        s += v.x * v.x + v.y * v.y + v.z * v.z + v.w * v.w;
    }
    norms[i] = s;
}

// ---------------------------------------------------------------------------
// Kernel 0b: split x into f16 hi/lo and store in WMMA-B register layout.
// Done ONCE, so the N^2 hot loop does zero conversions.
// ---------------------------------------------------------------------------
__global__ __launch_bounds__(256) void pairloss_pack(const float* __restrict__ x,
                                                     _Float16* __restrict__ bh,
                                                     _Float16* __restrict__ bl) {
    int idx = blockIdx.x * blockDim.x + threadIdx.x;   // over N*D elements
    if (idx >= NPTS * DIM) return;
    int col = idx / DIM;
    int k   = idx % DIM;
    float f = x[(size_t)col * DIM + k];
    _Float16 h = (_Float16)f;
    _Float16 l = (_Float16)(f - (float)h);
    size_t d = pack_idx(col, k);
    bh[d] = h;
    bl[d] = l;
}

// ---------------------------------------------------------------------------
// Kernel 1: fused Gram-tile + masked row reductions.
// Wave owns 16 rows (A in split-f16 registers, loaded once); WG = 4 waves.
// Per tile: 8 coalesced v16h loads + 12 WMMAs + minimal trans epilogue.
// partial layout: [(cs*NPTS + row)*4 + {P, Ng, PB, NB}]  (fully overwritten,
// no zero-init, no atomics -> deterministic).
// ---------------------------------------------------------------------------
__global__ __launch_bounds__(128) void pairloss_main(const float* __restrict__ x,
                                                     const _Float16* __restrict__ bh,
                                                     const _Float16* __restrict__ bl,
                                                     const int* __restrict__ tgt,
                                                     const float* __restrict__ norms,
                                                     float* __restrict__ partial,
                                                     int csplit) {
    const int lane  = threadIdx.x & 31;
    const int wave  = threadIdx.x >> 5;
    const int rb    = blockIdx.x / csplit;
    const int cs    = blockIdx.x % csplit;
    const int rowBase  = rb * 64 + wave * 16;
    const int colCount = NPTS / csplit;
    const int colStart = cs * colCount;

    const int hiSel = lane >> 4;   // 16-lane half-group
    const int l16   = lane & 15;

    // ---- A tile (16 rows x 128 K), split-f16, loaded once per wave ----
    // A operand in this code's contiguous k order (same as packed B): lane holds row M = lane%16;
    // halves i=0..15 -> K = 32c + 16*hiSel + i
    v16h a_hi[4], a_lo[4];
    const float* rowPtr = x + (size_t)(rowBase + l16) * DIM;
#pragma unroll
    for (int c = 0; c < 4; ++c) {
        int k0 = 32 * c + 16 * hiSel;
#pragma unroll
        for (int i = 0; i < 8; ++i) {
            float f = rowPtr[k0 + i];
            _Float16 h = (_Float16)f;
            a_hi[c][i] = h;
            a_lo[c][i] = (_Float16)(f - (float)h);
        }
#pragma unroll
        for (int i = 0; i < 8; ++i) {
            float f = rowPtr[k0 + 8 + i];
            _Float16 h = (_Float16)f;
            a_hi[c][8 + i] = h;
            a_lo[c][8 + i] = (_Float16)(f - (float)h);
        }
    }

    // C layout: lane holds column n = lane%16; vgpr r holds row (r + 8*hiSel)
    float sqRow[8];
    int   tRow[8];
#pragma unroll
    for (int r = 0; r < 8; ++r) {
        int rr = rowBase + r + 8 * hiSel;
        sqRow[r] = norms[rr];
        tRow[r]  = tgt[rr];
    }

    float accP[8], accN[8], accPB[8], accNB[8];
#pragma unroll
    for (int r = 0; r < 8; ++r) { accP[r] = accN[r] = accPB[r] = accNB[r] = 0.0f; }

    // exp() rewritten as raw v_exp_f32(fma(c1, d, c0)); args are in +-58,
    // far inside v_exp_f32's exact range (no fixup path reachable).
    const float EM_M = -ALPHA * L2E, EM_A = ALPHA * L2E;            // exp(A*(1-d))
    const float PB_M =  BETA * L2E,  PB_A = -0.8f * BETA * L2E;     // exp(B*(d-0.8))
    const float NB_M = -BETA * L2E,  NB_A =  1.1f * BETA * L2E;     // exp(B*(1.1-d))

    const v16h* bhv16 = (const v16h*)bh;
    const v16h* blv16 = (const v16h*)bl;

    for (int colBase = colStart; colBase < colStart + colCount; colBase += 16) {
        const int ct  = colBase >> 4;
        const int col = colBase + l16;

        const v16h* bhp = bhv16 + (size_t)ct * 128 + lane;   // 4 chunks * 32 lanes
        const v16h* blp = blv16 + (size_t)ct * 128 + lane;
        // prefetch next tile's packed data (global_prefetch_b8)
        __builtin_prefetch(bhp + 128, 0, 1);
        __builtin_prefetch(blp + 128, 0, 1);

        v8f acc = {0.f, 0.f, 0.f, 0.f, 0.f, 0.f, 0.f, 0.f};
#pragma unroll
        for (int c = 0; c < 4; ++c) {
            v16h bhc = bhp[c * 32];
            v16h blc = blp[c * 32];
            acc = __builtin_amdgcn_wmma_f32_16x16x32_f16(false, a_hi[c], false, bhc,
                                                         (short)0, acc, false, false);
            acc = __builtin_amdgcn_wmma_f32_16x16x32_f16(false, a_hi[c], false, blc,
                                                         (short)0, acc, false, false);
            acc = __builtin_amdgcn_wmma_f32_16x16x32_f16(false, a_lo[c], false, bhc,
                                                         (short)0, acc, false, false);
        }

        // ---- epilogue on C-layout registers ----
        float sqCol = norms[col];
        int   tCol  = tgt[col];
#pragma unroll
        for (int r = 0; r < 8; ++r) {
            int rr = rowBase + r + 8 * hiSel;
            float d2 = sqRow[r] + sqCol - 2.0f * acc[r];
            d2 = fmaxf(d2, 1e-12f);
            // d2 >= 1e-12 >> 2^-96: raw v_sqrt_f32, no rescale fixup needed
            float d = __builtin_amdgcn_sqrtf(d2);
            float em = __builtin_amdgcn_exp2f(fmaf(EM_M, d, EM_A));
            bool same = (tRow[r] == tCol);
            if (same) {
                if (rr != col) {   // pos_mask = same & ~eye
                    accP[r]  += em;
                    accPB[r] += __builtin_amdgcn_exp2f(fmaf(PB_M, d, PB_A));
                }
            } else {               // neg_mask = ~same
                accN[r]  += em;
                accNB[r] += __builtin_amdgcn_exp2f(fmaf(NB_M, d, NB_A));
            }
        }
    }

    // ---- reduce across the 16 lanes holding the same rows ----
#pragma unroll
    for (int r = 0; r < 8; ++r) {
#pragma unroll
        for (int m = 1; m < 16; m <<= 1) {
            accP[r]  += __shfl_xor(accP[r],  m, 16);
            accN[r]  += __shfl_xor(accN[r],  m, 16);
            accPB[r] += __shfl_xor(accPB[r], m, 16);
            accNB[r] += __shfl_xor(accNB[r], m, 16);
        }
    }
    if (l16 == 0) {
#pragma unroll
        for (int r = 0; r < 8; ++r) {
            int rr = rowBase + r + 8 * hiSel;
            float* p = partial + ((size_t)cs * NPTS + rr) * 4;
            p[0] = accP[r];
            p[1] = accN[r];
            p[2] = accPB[r];
            p[3] = accNB[r];
        }
    }
}

// ---------------------------------------------------------------------------
// Kernel 2: finalize. Sum csplit partials per row, per-row loss, mean -> out.
// Single block, deterministic tree reduction.
// ---------------------------------------------------------------------------
__global__ __launch_bounds__(1024) void pairloss_finalize(const float* __restrict__ partial,
                                                          float* __restrict__ out,
                                                          int csplit) {
    __shared__ float red[1024];
    const float LN2 = 0.6931471805599453f;
    float local = 0.0f;
    for (int r = threadIdx.x; r < NPTS; r += 1024) {
        float P = 0.f, Ng = 0.f, PB = 0.f, NB = 0.f;
        for (int cs = 0; cs < csplit; ++cs) {
            const float* p = partial + ((size_t)cs * NPTS + r) * 4;
            P  += p[0];
            Ng += p[1];
            PB += p[2];
            NB += p[3];
        }
        float a_lr = 1.0f - P / (P + Ng);
        // log(x) = log2(x) * ln2 via raw v_log_f32 (args are large positive sums)
        local += a_lr * ((__builtin_amdgcn_logf(PB) + __builtin_amdgcn_logf(NB)) * LN2);
    }
    red[threadIdx.x] = local;
    __syncthreads();
    for (int s = 512; s > 0; s >>= 1) {
        if ((int)threadIdx.x < s) red[threadIdx.x] += red[threadIdx.x + s];
        __syncthreads();
    }
    if (threadIdx.x == 0) out[0] = red[0] / (float)NPTS;
}

// ---------------------------------------------------------------------------
extern "C" void kernel_launch(void* const* d_in, const int* in_sizes, int n_in,
                              void* d_out, int out_size, void* d_ws, size_t ws_size,
                              hipStream_t stream) {
    const float* x   = (const float*)d_in[0];
    const int*   tgt = (const int*)d_in[1];
    float*       out = (float*)d_out;

    // workspace: bh[N*D] f16 | bl[N*D] f16 | norms[N] f32 | partial[csplit][N][4] f32
    _Float16* bh    = (_Float16*)d_ws;
    _Float16* bl    = bh + (size_t)NPTS * DIM;
    float*    norms = (float*)(bl + (size_t)NPTS * DIM);
    float*    partial = norms + NPTS;

    size_t fixedBytes = (size_t)NPTS * DIM * 2 * sizeof(_Float16) + NPTS * sizeof(float);
    int csplit = 8;
    while (csplit > 1 &&
           fixedBytes + 4ull * (size_t)csplit * NPTS * sizeof(float) > ws_size) {
        csplit >>= 1;
    }

    pairloss_norms<<<NPTS / 256, 256, 0, stream>>>(x, norms);
    pairloss_pack<<<(NPTS * DIM) / 256, 256, 0, stream>>>(x, bh, bl);

    int rowBlocks = NPTS / 64;   // 64 rows per WG (4 waves x 16 rows)
    pairloss_main<<<rowBlocks * csplit, 128, 0, stream>>>(x, bh, bl, tgt, norms,
                                                          partial, csplit);

    pairloss_finalize<<<1, 1024, 0, stream>>>(partial, out, csplit);
}
